// MLP_Q_45346264711716
// MI455X (gfx1250) — matrix operand order repaired and checked by reading, not compiled, not run on hardware
//
#include <hip/hip_runtime.h>
#include <hip/hip_fp16.h>

// ---------------------------------------------------------------------------
// Types
// ---------------------------------------------------------------------------
typedef _Float16 half_t;
typedef __attribute__((ext_vector_type(16))) _Float16 v16h;
typedef __attribute__((ext_vector_type(8)))  _Float16 v8h;
typedef __attribute__((ext_vector_type(4)))  _Float16 v4h;
typedef __attribute__((ext_vector_type(8)))  float    v8f;
typedef __attribute__((ext_vector_type(4)))  int      v4i;

union V16 { v16h v; v8h h[2]; };

// ---------------------------------------------------------------------------
// Problem constants (reference: E=2048, H=4E=8192, B*S=8192)
// ---------------------------------------------------------------------------
#define DIM_E 2048
#define DIM_H 8192
#define DIM_M 8192

// Tiling: block 128x256, 8 wave32 in 2x4 grid, wave tile 64x64 (4x4 WMMA).
#define BM 128
#define BN 256
#define BK 32
#define LDSS 40                     // LDS row stride in halves (32+8 pad)
#define NTHREADS 256
#define ABUF_BYTES (BM * LDSS * 2)  // 10240 B per A buffer
#define BBUF_BYTES (BN * LDSS * 2)  // 20480 B per B buffer

// ---------------------------------------------------------------------------
// CDNA5 async global->LDS copy (ASYNCcnt path, ISA 15.18.3 op 98)
// GV addressing: vdst = LDS byte address VGPR, vaddr = 64-bit global address.
// ---------------------------------------------------------------------------
__device__ __forceinline__ void async_copy_b128(unsigned lds_off, const void* gaddr) {
  asm volatile("global_load_async_to_lds_b128 %0, %1, off"
               :: "v"(lds_off), "v"(gaddr)
               : "memory");
}
__device__ __forceinline__ void wait_async0() {
  asm volatile("s_wait_asynccnt 0x0" ::: "memory");
}
// Generic LDS pointer -> wave-relative LDS byte offset (low 32 bits, ISA 10.2)
__device__ __forceinline__ unsigned lds_off32(const void* p) {
  return (unsigned)(unsigned long long)p;
}

// ---------------------------------------------------------------------------
// Elementwise conversion kernels (4 elements / thread, vectorized)
// ---------------------------------------------------------------------------
__global__ __launch_bounds__(256) void cvt_f32_to_f16_x4(
    const float* __restrict__ in, half_t* __restrict__ out) {
  size_t i = (size_t)blockIdx.x * blockDim.x + threadIdx.x;
  const float4 f = ((const float4*)in)[i];
  v4h o = { (half_t)f.x, (half_t)f.y, (half_t)f.z, (half_t)f.w };
  ((v4h*)out)[i] = o;
}

// w_f16 = (q - z); values in [-254,254] are exact in f16.
__global__ __launch_bounds__(256) void cvt_i32_to_f16_x4(
    const int* __restrict__ in, const int* __restrict__ zp,
    half_t* __restrict__ out) {
  size_t i = (size_t)blockIdx.x * blockDim.x + threadIdx.x;
  const v4i q = ((const v4i*)in)[i];
  const int z = zp[0];
  v4h o = { (half_t)(q.x - z), (half_t)(q.y - z),
            (half_t)(q.z - z), (half_t)(q.w - z) };
  ((v4h*)out)[i] = o;
}

// ---------------------------------------------------------------------------
// WMMA GEMM:  C[m,n] = f( s_w * sum_k A[m,k]*Bt[n,k]  +  s_b*(bias_q[n]-z_b) )
//   A : M x K row-major f16 (K contiguous),  Bt: N x K row-major f16
//   f = exact GELU (stage 1) or identity (stage 2)
// Block tile 128x256, BK=32; wave tile 64x64 -> 16 WMMA per 16 ds_load_b128
// (1.0 fragment load per WMMA). Global->LDS fills via CDNA5 async DMA,
// double-buffered, one barrier per K-step.
// ---------------------------------------------------------------------------
template<bool GELU, typename OutT>
__global__ __launch_bounds__(NTHREADS) void gemm_wmma_f16(
    const half_t* __restrict__ A,
    const half_t* __restrict__ Bt,
    OutT* __restrict__ C,
    const int*   __restrict__ bias_q,
    const float* __restrict__ s_w_p,
    const float* __restrict__ s_b_p,
    const int*   __restrict__ z_b_p,
    int M, int N, int K) {

  __shared__ half_t Asm[2][BM * LDSS];   // 2 x 10240 B
  __shared__ half_t Bsm[2][BN * LDSS];   // 2 x 20480 B

  const int tid  = threadIdx.x;
  const int lane = tid & 31;
  const int wave = tid >> 5;
  const int hs   = lane >> 4;      // half-wave select
  const int nl   = lane & 15;

  const int mBase = blockIdx.y * BM;
  const int nBase = blockIdx.x * BN;
  const int mW = (wave >> 2) * 64; // wave tile origin inside block tile
  const int nW = (wave & 3) * 64;

  // Global->LDS copy plan: 16B chunks, 4 chunks per 32-half row.
  // A: 128 rows * 4 = 512 chunks (2/thread). B: 256 rows * 4 = 1024 (4/thread).
  const char* gA[2]; unsigned la[2];
  const char* gB[4]; unsigned lb[4];
#pragma unroll
  for (int i = 0; i < 2; ++i) {
    const int c = tid + 256 * i, row = c >> 2, kc = c & 3;
    gA[i] = (const char*)(A + (size_t)(mBase + row) * K + kc * 8);
    la[i] = lds_off32(&Asm[0][row * LDSS + kc * 8]);
  }
#pragma unroll
  for (int i = 0; i < 4; ++i) {
    const int c = tid + 256 * i, row = c >> 2, kc = c & 3;
    gB[i] = (const char*)(Bt + (size_t)(nBase + row) * K + kc * 8);
    lb[i] = lds_off32(&Bsm[0][row * LDSS + kc * 8]);
  }

  // ---- stage K-tile 0 into buffer 0 (async DMA) ----
#pragma unroll
  for (int i = 0; i < 2; ++i) async_copy_b128(la[i], gA[i]);
#pragma unroll
  for (int i = 0; i < 4; ++i) async_copy_b128(lb[i], gB[i]);
  wait_async0();
  __syncthreads();

  v8f acc[4][4] = {};

  const int KT = K / BK;
  for (int kt = 0; kt < KT; ++kt) {
    // Kick off async fill of the next buffer; overlaps with WMMA below.
    // (Safe: last reader of that buffer was iteration kt-1, separated by
    //  the barrier at the end of iteration kt-1.)
    if (kt + 1 < KT) {
      const int boff = (kt + 1) * (BK * 2);          // byte advance along K
      const unsigned nb = (unsigned)((kt + 1) & 1);
#pragma unroll
      for (int i = 0; i < 2; ++i)
        async_copy_b128(la[i] + nb * ABUF_BYTES, gA[i] + boff);
#pragma unroll
      for (int i = 0; i < 4; ++i)
        async_copy_b128(lb[i] + nb * BBUF_BYTES, gB[i] + boff);
    }

    const int buf = kt & 1;

    // A fragments (ISA 16-bit A 16x32 layout):
    //   lanes 0-15 : K 0-7 then 16-23 ; lanes 16-31 : K 8-15 then 24-31
    V16 a[4];
#pragma unroll
    for (int mi = 0; mi < 4; ++mi) {
      const half_t* ap = &Asm[buf][(mW + mi * 16 + nl) * LDSS + 8 * hs];
      a[mi].h[0] = *(const v8h*)(ap);
      a[mi].h[1] = *(const v8h*)(ap + 16);
    }
    // B fragments (N x K striped): lane holds col n=l&15, K 8*(l>>4)..+7 then 16+8*(l>>4)..+7
    V16 b[4];
#pragma unroll
    for (int ni = 0; ni < 4; ++ni) {
      const half_t* bp = &Bsm[buf][(nW + ni * 16 + nl) * LDSS + 8 * hs];
      b[ni].h[0] = *(const v8h*)(bp);
      b[ni].h[1] = *(const v8h*)(bp + 16);
    }

#pragma unroll
    for (int mi = 0; mi < 4; ++mi)
#pragma unroll
      for (int ni = 0; ni < 4; ++ni)
        acc[mi][ni] = __builtin_amdgcn_wmma_f32_16x16x32_f16(
            false, a[mi].v, false, b[ni].v,
            (short)0, acc[mi][ni], false, false);

    // My async fills for the next buffer must land before anyone reads it.
    wait_async0();
    __syncthreads();
  }

  // ---- epilogue: scale + bias (+ exact GELU), write out ----
  const float s_w = s_w_p[0];
  const float s_b = s_b_p[0];
  const int   z_b = z_b_p[0];

#pragma unroll
  for (int mi = 0; mi < 4; ++mi) {
#pragma unroll
    for (int ni = 0; ni < 4; ++ni) {
      const int n = nBase + nW + ni * 16 + nl;
      const float bv = (float)(bias_q[n] - z_b) * s_b;
#pragma unroll
      for (int r = 0; r < 8; ++r) {
        const int m = mBase + mW + mi * 16 + 8 * hs + r;  // C/D VGPR layout
        float v = acc[mi][ni][r] * s_w + bv;
        if (GELU)
          v = 0.5f * v * (1.0f + erff(v * 0.70710678118654752f));
        C[(size_t)m * N + n] = (OutT)v;
      }
    }
  }
}

// ---------------------------------------------------------------------------
// Host launcher
// ---------------------------------------------------------------------------
extern "C" void kernel_launch(void* const* d_in, const int* in_sizes, int n_in,
                              void* d_out, int out_size, void* d_ws, size_t ws_size,
                              hipStream_t stream) {
  (void)in_sizes; (void)n_in; (void)out_size; (void)ws_size;

  const float* x        = (const float*)d_in[0];
  const int*   w_fc_q   = (const int*)  d_in[1];
  const int*   b_fc_q   = (const int*)  d_in[2];
  const int*   w_proj_q = (const int*)  d_in[3];
  const int*   b_proj_q = (const int*)  d_in[4];
  const float* s_fc_w   = (const float*)d_in[5];
  const float* s_fc_b   = (const float*)d_in[6];
  const float* s_proj_w = (const float*)d_in[7];
  const float* s_proj_b = (const float*)d_in[8];
  const int*   z_fc_w   = (const int*)  d_in[9];
  const int*   z_fc_b   = (const int*)  d_in[10];
  const int*   z_proj_w = (const int*)  d_in[11];
  const int*   z_proj_b = (const int*)  d_in[12];

  // Workspace layout (halves): xh | wfc | wproj | h
  half_t* xh     = (half_t*)d_ws;                                   // M*E
  half_t* wfch   = xh     + (size_t)DIM_M * DIM_E;                  // H*E
  half_t* wprojh = wfch   + (size_t)DIM_H * DIM_E;                  // E*H
  half_t* hbuf   = wprojh + (size_t)DIM_E * DIM_H;                  // M*H

  // 1) conversions (16,777,216 elements each -> 16384 blocks of 256x4)
  const int cvtBlocks = (DIM_M * DIM_E) / (256 * 4);                // 16384
  cvt_f32_to_f16_x4<<<cvtBlocks, 256, 0, stream>>>(x, xh);
  cvt_i32_to_f16_x4<<<(DIM_H * DIM_E) / (256 * 4), 256, 0, stream>>>(
      w_fc_q, z_fc_w, wfch);
  cvt_i32_to_f16_x4<<<(DIM_E * DIM_H) / (256 * 4), 256, 0, stream>>>(
      w_proj_q, z_proj_w, wprojh);

  // 2) GEMM1 + exact GELU:  h[M,H] = gelu(s_w * x·w_fcᵀ + s_b*(b_fc - z))
  {
    dim3 grid(DIM_H / BN, DIM_M / BM);   // 32 x 64
    gemm_wmma_f16<true, half_t><<<grid, NTHREADS, 0, stream>>>(
        xh, wfch, hbuf, b_fc_q, s_fc_w, s_fc_b, z_fc_b,
        DIM_M, DIM_H, DIM_E);
  }

  // 3) GEMM2: out[M,E] = s_w * h·w_projᵀ + s_b*(b_proj - z)
  {
    dim3 grid(DIM_E / BN, DIM_M / BM);   // 8 x 64
    gemm_wmma_f16<false, float><<<grid, NTHREADS, 0, stream>>>(
        hbuf, wprojh, (float*)d_out, b_proj_q, s_proj_w, s_proj_b, z_proj_b,
        DIM_M, DIM_E, DIM_H);
  }
}
